// AttentionBlock_73272142070433
// MI455X (gfx1250) — matrix operand order repaired and checked by reading, not compiled, not run on hardware
//
#include <hip/hip_runtime.h>

#define Bn  32
#define Cd  256
#define Sd  1024
#define Gn  32
#define CPG 8

typedef __attribute__((ext_vector_type(16))) __bf16   v16bf;
typedef __attribute__((ext_vector_type(8)))  float    v8f;
typedef __attribute__((ext_vector_type(4)))  unsigned u32x4;

__device__ __forceinline__ __bf16 f2bf(float f) {
    unsigned u = __builtin_bit_cast(unsigned, f);
    unsigned r = (u + 0x7FFFu + ((u >> 16) & 1u)) >> 16;  // round-to-nearest-even
    unsigned short s = (unsigned short)r;
    return __builtin_bit_cast(__bf16, s);
}

__device__ __forceinline__ v8f zero8() {
    v8f r;
    for (int i = 0; i < 8; ++i) r[i] = 0.0f;
    return r;
}

// A-operand (16x32 bf16, M x K): lanes 0-15 hold M=lane, K={0..7,16..23};
// lanes 16-31 hold M=lane-16, K={8..15,24..31}. Row-major A, K contiguous.
__device__ __forceinline__ v16bf load_frag_a(const __bf16* A, int lda, int m0, int k0, int lane) {
    int hl = lane & 15, hi = lane >> 4;
    const u32x4* p = (const u32x4*)(A + (size_t)(m0 + hl) * lda + k0);
    union { v16bf v; u32x4 q[2]; } u;
    u.q[0] = p[hi];        // K = k0 + hi*8 .. +7
    u.q[1] = p[2 + hi];    // K = k0 + 16 + hi*8 .. +7
    return u.v;
}

// B-operand (32x16 bf16, K x N) loaded from N-major storage Bt[N][K]:
// lanes 0-15: N=lane, K={0..7,16..23}; lanes 16-31: N=lane-16, K={8..15,24..31}.
__device__ __forceinline__ v16bf load_frag_b(const __bf16* Bt, int ldb, int n0, int k0, int lane) {
    int hl = lane & 15, hi = lane >> 4;
    const u32x4* p = (const u32x4*)(Bt + (size_t)(n0 + hl) * ldb + k0);
    union { v16bf v; u32x4 q[2]; } u;
    u.q[0] = p[hi];
    u.q[1] = p[2 + hi];
    return u.v;
}

__device__ __forceinline__ v8f wmma_bf16(v16bf a, v16bf b, v8f c) {
    return __builtin_amdgcn_wmma_f32_16x16x32_bf16(false, a, false, b, (short)0, c, false, false);
}

// ---------------- Kernel 1: GroupNorm statistics ----------------
__global__ __launch_bounds__(256) void gn_stats_kernel(const float* __restrict__ x,
                                                       float* __restrict__ mean,
                                                       float* __restrict__ rstd) {
    int bg = blockIdx.x;                         // b*Gn + g ; group channels contiguous
    const float* p = x + (size_t)bg * (CPG * Sd);
    float s1 = 0.f, s2 = 0.f;
    for (int i = threadIdx.x; i < CPG * Sd; i += 256) {
        float v = p[i];
        s1 += v; s2 += v * v;
    }
    __shared__ float r1[256], r2[256];
    r1[threadIdx.x] = s1; r2[threadIdx.x] = s2;
    __syncthreads();
    for (int off = 128; off > 0; off >>= 1) {
        if (threadIdx.x < off) {
            r1[threadIdx.x] += r1[threadIdx.x + off];
            r2[threadIdx.x] += r2[threadIdx.x + off];
        }
        __syncthreads();
    }
    if (threadIdx.x == 0) {
        float m = r1[0] * (1.0f / (CPG * Sd));
        float var = r2[0] * (1.0f / (CPG * Sd)) - m * m;
        mean[bg] = m;
        rstd[bg] = rsqrtf(var + 1e-5f);
    }
}

// ---------------- Kernel 2: weight convert + transpose (f32[Cin,Cout] -> bf16[Cout,Cin]) ---
__global__ __launch_bounds__(256) void wconv_kernel(const float* __restrict__ W,
                                                    __bf16* __restrict__ Wt) {
    int j = blockIdx.x * 256 + threadIdx.x;      // j = co*Cd + ci (coalesced store)
    int co = j >> 8, ci = j & (Cd - 1);
    Wt[j] = f2bf(W[(size_t)ci * Cd + co]);
}

// ---------------- Kernel 3: normalize + transpose x[B,C,S] -> h bf16 [B,S,C] ----------------
__global__ __launch_bounds__(256) void norm_transpose_kernel(const float* __restrict__ x,
                                                             const float* __restrict__ mean,
                                                             const float* __restrict__ rstd,
                                                             __bf16* __restrict__ h) {
    __shared__ float tile[32][33];
    int b  = blockIdx.z;
    int c0 = blockIdx.y * 32;
    int s0 = blockIdx.x * 32;
    int tid = threadIdx.x;
    int sj = tid & 31, ci = tid >> 5;            // read: coalesced along S
    for (int r = 0; r < 4; ++r) {
        int c = c0 + ci + r * 8;
        int g = b * Gn + (c >> 3);
        float v = x[(size_t)b * Cd * Sd + (size_t)c * Sd + s0 + sj];
        tile[ci + r * 8][sj] = (v - mean[g]) * rstd[g];
    }
    __syncthreads();
    int cj = tid & 31, si = tid >> 5;            // write: coalesced along C
    for (int r = 0; r < 4; ++r) {
        int s = s0 + si + r * 8;
        h[(size_t)b * Sd * Cd + (size_t)s * Cd + c0 + cj] = f2bf(tile[cj][si + r * 8]);
    }
}

// ---------------- Kernel 4: WMMA GEMM  out = h @ W + bias  (tile 128x64, 8 waves) -------
__global__ __launch_bounds__(256) void gemm_qkv_kernel(const __bf16* __restrict__ A,
                                                       const __bf16* __restrict__ Bt,
                                                       const float* __restrict__ bias,
                                                       __bf16* __restrict__ out,
                                                       int transposed) {
    int bz = blockIdx.z;
    int m0 = blockIdx.y * 128;
    int n0 = blockIdx.x * 64;
    int tid = threadIdx.x, lane = tid & 31, wid = tid >> 5;
    int wm = wid & 3, wn = wid >> 2;
    int hl = lane & 15, hi = lane >> 4;
    const __bf16* Ab = A + (size_t)bz * Sd * Cd;
    int mBase = m0 + wm * 32, nBase = n0 + wn * 32;

    v8f acc[2][2];
    acc[0][0] = zero8(); acc[0][1] = zero8(); acc[1][0] = zero8(); acc[1][1] = zero8();

    for (int ks = 0; ks < Cd; ks += 32) {
        if (ks + 32 < Cd) {
            __builtin_prefetch(Ab + (size_t)(mBase + hl) * Cd + ks + 32, 0, 1);
            __builtin_prefetch(Bt + (size_t)(nBase + hl) * Cd + ks + 32, 0, 1);
        }
        v16bf a0 = load_frag_a(Ab, Cd, mBase,      ks, lane);
        v16bf a1 = load_frag_a(Ab, Cd, mBase + 16, ks, lane);
        v16bf b0 = load_frag_b(Bt, Cd, nBase,      ks, lane);
        v16bf b1 = load_frag_b(Bt, Cd, nBase + 16, ks, lane);
        acc[0][0] = wmma_bf16(a0, b0, acc[0][0]);
        acc[0][1] = wmma_bf16(a0, b1, acc[0][1]);
        acc[1][0] = wmma_bf16(a1, b0, acc[1][0]);
        acc[1][1] = wmma_bf16(a1, b1, acc[1][1]);
    }
    for (int i = 0; i < 2; ++i)
        for (int j = 0; j < 2; ++j)
            for (int r = 0; r < 8; ++r) {
                int m = mBase + i * 16 + hi * 8 + r;
                int n = nBase + j * 16 + hl;
                float val = acc[i][j][r] + bias[n];
                if (!transposed)
                    out[(size_t)bz * Sd * Cd + (size_t)m * Cd + n] = f2bf(val);  // [S,C]
                else
                    out[(size_t)bz * Cd * Sd + (size_t)n * Sd + m] = f2bf(val);  // [C,S] (V^T)
            }
}

// ---------------- Kernel 5: attention (32 queries per WG, f32 scores in 128KB LDS) --------
__global__ __launch_bounds__(256) void attn_kernel(const __bf16* __restrict__ Q,
                                                   const __bf16* __restrict__ Km,
                                                   const __bf16* __restrict__ Vt,
                                                   __bf16* __restrict__ O,
                                                   float scale) {
    extern __shared__ float smem[];
    float* sc  = smem;                 // [32][1024] scores/probs
    float* red = smem + 32 * 1024;     // [32][8] reduction scratch
    int b  = blockIdx.y;
    int qr0 = blockIdx.x * 32;
    int tid = threadIdx.x, lane = tid & 31, wid = tid >> 5;
    int hl = lane & 15, hi = lane >> 4;
    const __bf16* Qb = Q  + (size_t)b * Sd * Cd;
    const __bf16* Kb = Km + (size_t)b * Sd * Cd;
    const __bf16* Vb = Vt + (size_t)b * Cd * Sd;

    int rg = wid & 1;                  // row group: 16 rows
    int cq = wid >> 1;                 // column quarter: 256 keys

    // ---- Phase 1: S = scale * Q @ K^T (Q fragments held resident) ----
    v16bf qf[8];
    for (int ks = 0; ks < 8; ++ks)
        qf[ks] = load_frag_a(Qb, Cd, qr0 + rg * 16, ks * 32, lane);
    for (int nc = 0; nc < 8; ++nc) {
        int n0 = cq * 256 + nc * 32;
        v8f acc0 = zero8(), acc1 = zero8();
        for (int ks = 0; ks < 8; ++ks) {
            v16bf b0 = load_frag_b(Kb, Cd, n0,      ks * 32, lane);
            v16bf b1 = load_frag_b(Kb, Cd, n0 + 16, ks * 32, lane);
            acc0 = wmma_bf16(qf[ks], b0, acc0);
            acc1 = wmma_bf16(qf[ks], b1, acc1);
        }
        for (int r = 0; r < 8; ++r) {
            int m = rg * 16 + hi * 8 + r;
            sc[m * 1024 + n0 + hl]      = acc0[r] * scale;
            sc[m * 1024 + n0 + 16 + hl] = acc1[r] * scale;
        }
    }
    __syncthreads();

    // ---- Phase 2: softmax rows in f32 (8 threads per row) ----
    int row = tid >> 3, seg = tid & 7;
    float* rp = sc + row * 1024;
    float mx = -3.4e38f;
    for (int i = seg * 128; i < seg * 128 + 128; ++i) mx = fmaxf(mx, rp[i]);
    red[row * 8 + seg] = mx;
    __syncthreads();
    mx = red[row * 8];
    for (int i = 1; i < 8; ++i) mx = fmaxf(mx, red[row * 8 + i]);
    __syncthreads();
    float sum = 0.f;
    for (int i = seg * 128; i < seg * 128 + 128; ++i) {
        float e = __expf(rp[i] - mx);
        rp[i] = e; sum += e;
    }
    red[row * 8 + seg] = sum;
    __syncthreads();
    sum = 0.f;
    for (int i = 0; i < 8; ++i) sum += red[row * 8 + i];
    float inv = 1.0f / sum;
    for (int i = seg * 128; i < seg * 128 + 128; ++i) rp[i] *= inv;
    __syncthreads();

    // ---- Phase 3: O = P @ V (P from LDS, V^T contiguous) ----
    v8f oacc[4];
    for (int j = 0; j < 4; ++j) oacc[j] = zero8();
    for (int ks = 0; ks < 32; ++ks) {
        // Build A fragment of P from LDS (f32 -> bf16, ISA A layout)
        const float* prow = sc + (rg * 16 + hl) * 1024 + ks * 32;
        v16bf pa;
        for (int j = 0; j < 8; ++j) pa[j]     = f2bf(prow[hi * 8 + j]);
        for (int j = 0; j < 8; ++j) pa[8 + j] = f2bf(prow[16 + hi * 8 + j]);
        for (int j = 0; j < 4; ++j) {
            v16bf vb = load_frag_b(Vb, Sd, cq * 64 + j * 16, ks * 32, lane);
            oacc[j] = wmma_bf16(pa, vb, oacc[j]);
        }
    }
    for (int j = 0; j < 4; ++j)
        for (int r = 0; r < 8; ++r) {
            int srow = qr0 + rg * 16 + hi * 8 + r;
            int c    = cq * 64 + j * 16 + hl;
            O[(size_t)b * Sd * Cd + (size_t)srow * Cd + c] = f2bf(oacc[j][r]);
        }
}

// ---------------- Kernel 6: out = x + (O @ Wo + bo), stored as [B,C,S] f32 ----------------
__global__ __launch_bounds__(256) void proj_residual_kernel(const __bf16* __restrict__ A,
                                                            const __bf16* __restrict__ Bt,
                                                            const float* __restrict__ bias,
                                                            const float* __restrict__ xres,
                                                            float* __restrict__ out) {
    int bz = blockIdx.z;
    int m0 = blockIdx.y * 128;
    int n0 = blockIdx.x * 64;
    int tid = threadIdx.x, lane = tid & 31, wid = tid >> 5;
    int wm = wid & 3, wn = wid >> 2;
    int hl = lane & 15, hi = lane >> 4;
    const __bf16* Ab = A + (size_t)bz * Sd * Cd;
    int mBase = m0 + wm * 32, nBase = n0 + wn * 32;

    v8f acc[2][2];
    acc[0][0] = zero8(); acc[0][1] = zero8(); acc[1][0] = zero8(); acc[1][1] = zero8();

    for (int ks = 0; ks < Cd; ks += 32) {
        v16bf a0 = load_frag_a(Ab, Cd, mBase,      ks, lane);
        v16bf a1 = load_frag_a(Ab, Cd, mBase + 16, ks, lane);
        v16bf b0 = load_frag_b(Bt, Cd, nBase,      ks, lane);
        v16bf b1 = load_frag_b(Bt, Cd, nBase + 16, ks, lane);
        acc[0][0] = wmma_bf16(a0, b0, acc[0][0]);
        acc[0][1] = wmma_bf16(a0, b1, acc[0][1]);
        acc[1][0] = wmma_bf16(a1, b0, acc[1][0]);
        acc[1][1] = wmma_bf16(a1, b1, acc[1][1]);
    }
    for (int i = 0; i < 2; ++i)
        for (int j = 0; j < 2; ++j)
            for (int r = 0; r < 8; ++r) {
                int m = mBase + i * 16 + hi * 8 + r;   // token index s
                int n = nBase + j * 16 + hl;           // channel c
                size_t oidx = (size_t)bz * Cd * Sd + (size_t)n * Sd + m;
                out[oidx] = xres[oidx] + acc[i][j][r] + bias[n];
            }
}

extern "C" void kernel_launch(void* const* d_in, const int* in_sizes, int n_in,
                              void* d_out, int out_size, void* d_ws, size_t ws_size,
                              hipStream_t stream) {
    const float* x  = (const float*)d_in[0];
    const float* WQ = (const float*)d_in[1];
    const float* bQ = (const float*)d_in[2];
    const float* WK = (const float*)d_in[3];
    const float* bK = (const float*)d_in[4];
    const float* WV = (const float*)d_in[5];
    const float* bV = (const float*)d_in[6];
    const float* Wo = (const float*)d_in[7];
    const float* bo = (const float*)d_in[8];
    float* out = (float*)d_out;

    char* ws = (char*)d_ws;
    size_t off = 0;
    auto alloc = [&](size_t bytes) -> char* {
        char* p = ws + off;
        off += (bytes + 255) & ~(size_t)255;
        return p;
    };
    float*  mean = (float*)alloc((size_t)Bn * Gn * 4);
    float*  rstd = (float*)alloc((size_t)Bn * Gn * 4);
    __bf16* WQt  = (__bf16*)alloc((size_t)Cd * Cd * 2);
    __bf16* WKt  = (__bf16*)alloc((size_t)Cd * Cd * 2);
    __bf16* WVt  = (__bf16*)alloc((size_t)Cd * Cd * 2);
    __bf16* WOt  = (__bf16*)alloc((size_t)Cd * Cd * 2);
    __bf16* h    = (__bf16*)alloc((size_t)Bn * Sd * Cd * 2);
    __bf16* Qm   = (__bf16*)alloc((size_t)Bn * Sd * Cd * 2);
    __bf16* Km   = (__bf16*)alloc((size_t)Bn * Sd * Cd * 2);
    __bf16* Vt   = (__bf16*)alloc((size_t)Bn * Sd * Cd * 2);
    __bf16* Om   = (__bf16*)alloc((size_t)Bn * Sd * Cd * 2);

    gn_stats_kernel<<<Bn * Gn, 256, 0, stream>>>(x, mean, rstd);

    wconv_kernel<<<(Cd * Cd) / 256, 256, 0, stream>>>(WQ, WQt);
    wconv_kernel<<<(Cd * Cd) / 256, 256, 0, stream>>>(WK, WKt);
    wconv_kernel<<<(Cd * Cd) / 256, 256, 0, stream>>>(WV, WVt);
    wconv_kernel<<<(Cd * Cd) / 256, 256, 0, stream>>>(Wo, WOt);

    norm_transpose_kernel<<<dim3(Sd / 32, Cd / 32, Bn), 256, 0, stream>>>(x, mean, rstd, h);

    dim3 gGemm(Cd / 64, Sd / 128, Bn);
    gemm_qkv_kernel<<<gGemm, 256, 0, stream>>>(h, WQt, bQ, Qm, 0);
    gemm_qkv_kernel<<<gGemm, 256, 0, stream>>>(h, WKt, bK, Km, 0);
    gemm_qkv_kernel<<<gGemm, 256, 0, stream>>>(h, WVt, bV, Vt, 1);  // store V^T [C,S]

    size_t shmem = (size_t)(32 * 1024 + 32 * 8) * sizeof(float);    // 132 KB of 320 KB LDS
    attn_kernel<<<dim3(Sd / 32, Bn), 256, shmem, stream>>>(Qm, Km, Vt, Om, 0.0625f); // 256^-0.5

    proj_residual_kernel<<<gGemm, 256, 0, stream>>>(Om, WOt, bo, x, out);
}
